// MultiCrossAttention_37014028157470
// MI455X (gfx1250) — compile-verified, matrix operand order checked by reading, not run on hardware
//
#include <hip/hip_runtime.h>

typedef unsigned short ushort_t;
typedef unsigned int   uint_t;
typedef __attribute__((ext_vector_type(16))) __bf16 v16bf;
typedef __attribute__((ext_vector_type(8)))  float  v8f;
typedef __attribute__((ext_vector_type(4)))  int    v4i;

union FragBF { v16bf v; uint4 q[2]; };

__device__ __forceinline__ ushort_t f2bf(float f) {
  uint_t u = __builtin_bit_cast(uint_t, f);
  u += 0x7FFFu + ((u >> 16) & 1u);      // round-to-nearest-even
  return (ushort_t)(u >> 16);
}

constexpr int cB = 4, cT = 2048, cC = 768, cH = 12, cHS = 64;

// ---------------------------------------------------------------------------
// Async global->LDS staging (gfx1250 GLOBAL_LOAD_ASYNC_TO_LDS_B128, ASYNCcnt).
// Builtin signature (from hipcc diagnostic): (v4i AS1*, v4i AS3*, Ii, Ii).
// Falls back to a VGPR round-trip if the builtin is unavailable.
// ---------------------------------------------------------------------------
#if defined(__has_builtin)
#if __has_builtin(__builtin_amdgcn_global_load_async_to_lds_b128) && \
    __has_builtin(__builtin_amdgcn_s_wait_asynccnt)
#define USE_ASYNC_LDS 1
#endif
#endif
#ifndef USE_ASYNC_LDS
#define USE_ASYNC_LDS 0
#endif

typedef __attribute__((address_space(1))) v4i* gptr_v4i;
typedef __attribute__((address_space(3))) v4i* lptr_v4i;

__device__ __forceinline__ void stage_b128(const ushort_t* g, ushort_t* l) {
#if USE_ASYNC_LDS
  __builtin_amdgcn_global_load_async_to_lds_b128(
      (gptr_v4i)g, (lptr_v4i)l, 0, 0);
#else
  *(uint4*)l = *(const uint4*)g;
#endif
}

__device__ __forceinline__ void stage_join() {
#if USE_ASYNC_LDS
  __builtin_amdgcn_s_wait_asynccnt(0);
#endif
}

// ---------------------------------------------------------------------------
// fp32 -> bf16 elementwise conversion
// ---------------------------------------------------------------------------
__global__ __launch_bounds__(256)
void cvt_f32_bf16(const float* __restrict__ s, ushort_t* __restrict__ d, int n) {
  int i = (blockIdx.x * 256 + threadIdx.x) * 4;
  if (i < n) {
    float4 f = *(const float4*)&s[i];
    uint2 u;
    u.x = (uint_t)f2bf(f.x) | ((uint_t)f2bf(f.y) << 16);
    u.y = (uint_t)f2bf(f.z) | ((uint_t)f2bf(f.w) << 16);
    *(uint2*)&d[i] = u;
  }
}

// ---------------------------------------------------------------------------
// 64x64 tile GEMM: C[M,N] = A[M,K](bf16,row) * B[K,N](bf16,row) + bias
// 128 threads = 4 waves; each wave owns a 32x32 sub-tile (2x2 WMMA tiles).
// MODE 0: f32 row-major out.  MODE 2: bf16 [B,H,T,HS] (Q).
// MODE 3: bf16 split K/V -> dst0/dst1, each [B,H,T,HS].
// ---------------------------------------------------------------------------
template<int MODE>
__global__ __launch_bounds__(128)
void gemm64(const ushort_t* __restrict__ A, const ushort_t* __restrict__ Bw,
            const float* __restrict__ bias, void* dst0, void* dst1,
            int M, int N, int K)
{
  (void)M;
  const int n0   = blockIdx.x * 64;
  const int m0   = blockIdx.y * 64;
  const int tid  = threadIdx.x;
  const int wave = tid >> 5, lane = tid & 31;
  const int wm   = (wave >> 1) * 32;
  const int wn   = (wave & 1) * 32;
  const int lr   = lane & 15;
  const int kb8  = (lane >> 4) * 8;       // == 8 * (lane >= 16)

  __shared__ alignas(16) ushort_t As[64 * 32];   // [m][k]
  __shared__ alignas(16) ushort_t Bt[64 * 32];   // [n][k] (transposed)

  v8f acc[2][2] = {};

  for (int kk = 0; kk < K; kk += 32) {
    __syncthreads();
    // stage A tile 64x32: contiguous -> async global->LDS
    #pragma unroll
    for (int it = 0; it < 2; ++it) {
      int idx = (tid + it * 128) * 8;
      int r = idx >> 5, c = idx & 31;
      stage_b128(&A[(size_t)(m0 + r) * K + kk + c], &As[idx]);
      if (kk + 32 < K)
        __builtin_prefetch(&A[(size_t)(m0 + r) * K + kk + 32 + c], 0, 0);
    }
    // stage B tile 32x64 transposed into [n][k] (VGPR path: scatter)
    #pragma unroll
    for (int it = 0; it < 2; ++it) {
      int idx = (tid + it * 128) * 8;
      int r = idx >> 6, c = idx & 63;
      uint4 d = *(const uint4*)&Bw[(size_t)(kk + r) * N + n0 + c];
      const ushort_t* p = (const ushort_t*)&d;
      #pragma unroll
      for (int j = 0; j < 8; ++j) Bt[(c + j) * 32 + r] = p[j];
    }
    stage_join();
    __syncthreads();

    FragBF af[2], bfr[2];
    #pragma unroll
    for (int mt = 0; mt < 2; ++mt) {
      int row = wm + mt * 16 + lr;
      af[mt].q[0] = *(const uint4*)&As[row * 32 + kb8];
      af[mt].q[1] = *(const uint4*)&As[row * 32 + kb8 + 16];
    }
    #pragma unroll
    for (int nt = 0; nt < 2; ++nt) {
      int col = wn + nt * 16 + lr;
      bfr[nt].q[0] = *(const uint4*)&Bt[col * 32 + kb8];
      bfr[nt].q[1] = *(const uint4*)&Bt[col * 32 + kb8 + 16];
    }
    #pragma unroll
    for (int mt = 0; mt < 2; ++mt)
      #pragma unroll
      for (int nt = 0; nt < 2; ++nt)
        acc[mt][nt] = __builtin_amdgcn_wmma_f32_16x16x32_bf16(
            false, af[mt].v, false, bfr[nt].v, (short)0, acc[mt][nt], false, false);
  }

  #pragma unroll
  for (int mt = 0; mt < 2; ++mt) {
    #pragma unroll
    for (int nt = 0; nt < 2; ++nt) {
      int n = n0 + wn + nt * 16 + lr;
      float bv = bias[n];
      #pragma unroll
      for (int r = 0; r < 8; ++r) {
        int m = m0 + wm + mt * 16 + r + kb8;  // C layout: M = r + 8*(lane>=16)
        float val = acc[mt][nt][r] + bv;
        if (MODE == 0) {
          ((float*)dst0)[(size_t)m * N + n] = val;
        } else if (MODE == 2) {
          int b = m / cT, t = m % cT;
          int h = n / cHS, hs = n % cHS;
          ((ushort_t*)dst0)[(((size_t)(b * cH + h)) * cT + t) * cHS + hs] = f2bf(val);
        } else if (MODE == 3) {
          int b = m / cT, t = m % cT;
          int n2 = n;
          ushort_t* dk = (ushort_t*)dst0;
          if (n2 >= cC) { n2 -= cC; dk = (ushort_t*)dst1; }
          int h = n2 / cHS, hs = n2 % cHS;
          dk[(((size_t)(b * cH + h)) * cT + t) * cHS + hs] = f2bf(val);
        }
      }
    }
  }
}

// ---------------------------------------------------------------------------
// Flash attention: one block per (b,h, 64-query tile); 4 waves, 16 rows each.
// S = Q K^T * scale, online softmax, O = P V, y = O / l  (bf16 out, [B*T,C]).
// ---------------------------------------------------------------------------
__global__ __launch_bounds__(128)
void attn64(const ushort_t* __restrict__ qh, const ushort_t* __restrict__ kh,
            const ushort_t* __restrict__ vh, ushort_t* __restrict__ yb)
{
  const int qb   = blockIdx.x;
  const int bh   = blockIdx.y;
  const int b    = bh / cH, h = bh % cH;
  const int tid  = threadIdx.x;
  const int wave = tid >> 5, lane = tid & 31;
  const int lr   = lane & 15;
  const int kb8  = (lane >> 4) * 8;

  __shared__ alignas(16) ushort_t Qs[64 * 64];     // [q][hs]
  __shared__ alignas(16) ushort_t Ks[64 * 64];     // [key][hs]
  __shared__ alignas(16) ushort_t Vt[64 * 64];     // [hs][key] (transposed)
  __shared__ alignas(16) ushort_t Ps[4][16 * 64];  // wave-private P tiles

  const ushort_t* qbase = qh + ((size_t)bh * cT + qb * 64) * cHS;
  const ushort_t* kbase = kh + (size_t)bh * cT * cHS;
  const ushort_t* vbase = vh + (size_t)bh * cT * cHS;

  #pragma unroll
  for (int it = 0; it < 4; ++it) {
    int idx = (tid + it * 128) * 8;
    stage_b128(&qbase[idx], &Qs[idx]);
  }
  stage_join();
  __syncthreads();

  FragBF aq[2];
  #pragma unroll
  for (int kf = 0; kf < 2; ++kf) {
    int base = kf * 32 + kb8;
    int row  = wave * 16 + lr;
    aq[kf].q[0] = *(const uint4*)&Qs[row * 64 + base];
    aq[kf].q[1] = *(const uint4*)&Qs[row * 64 + base + 16];
  }

  v8f o[4] = {};
  float mrun[8], lrun[8];
  #pragma unroll
  for (int r = 0; r < 8; ++r) { mrun[r] = -1e30f; lrun[r] = 0.0f; }

  const float scale = 0.125f;  // 1/sqrt(64)

  for (int kc = 0; kc < cT; kc += 64) {
    __syncthreads();  // previous iteration finished reading Ks/Vt
    // K tile: contiguous -> async global->LDS
    #pragma unroll
    for (int it = 0; it < 4; ++it) {
      int idx = (tid + it * 128) * 8;
      stage_b128(&kbase[kc * cHS + idx], &Ks[idx]);
    }
    // V tile: transposed scatter (VGPR path)
    #pragma unroll
    for (int it = 0; it < 4; ++it) {
      int idx = (tid + it * 128) * 8;
      int r = idx >> 6, c = idx & 63;
      uint4 d = *(const uint4*)&vbase[kc * cHS + idx];
      const ushort_t* p = (const ushort_t*)&d;
      #pragma unroll
      for (int j = 0; j < 8; ++j) Vt[(c + j) * 64 + r] = p[j];
    }
    stage_join();
    __syncthreads();

    // S tiles: 16 queries x 64 keys per wave
    v8f s[4] = {};
    #pragma unroll
    for (int kf = 0; kf < 2; ++kf) {
      int base = kf * 32 + kb8;
      #pragma unroll
      for (int nt = 0; nt < 4; ++nt) {
        FragBF bk;
        int col = nt * 16 + lr;  // key index
        bk.q[0] = *(const uint4*)&Ks[col * 64 + base];
        bk.q[1] = *(const uint4*)&Ks[col * 64 + base + 16];
        s[nt] = __builtin_amdgcn_wmma_f32_16x16x32_bf16(
            false, aq[kf].v, false, bk.v, (short)0, s[nt], false, false);
      }
    }

    // online softmax per row (row M = r + 8*(lane>=16), spread over 16 lanes)
    #pragma unroll
    for (int r = 0; r < 8; ++r) {
      float cm = -1e30f;
      #pragma unroll
      for (int nt = 0; nt < 4; ++nt) cm = fmaxf(cm, s[nt][r]);
      cm *= scale;
      #pragma unroll
      for (int msk = 1; msk < 16; msk <<= 1)
        cm = fmaxf(cm, __shfl_xor(cm, msk, 32));
      float nm    = fmaxf(mrun[r], cm);
      float alpha = __expf(mrun[r] - nm);
      float rs = 0.0f;
      #pragma unroll
      for (int nt = 0; nt < 4; ++nt) {
        float pv = __expf(s[nt][r] * scale - nm);
        s[nt][r] = pv;
        rs += pv;
      }
      #pragma unroll
      for (int msk = 1; msk < 16; msk <<= 1)
        rs += __shfl_xor(rs, msk, 32);
      lrun[r] = lrun[r] * alpha + rs;
      mrun[r] = nm;
      #pragma unroll
      for (int j = 0; j < 4; ++j) o[j][r] *= alpha;
    }

    // C-layout -> A-layout via wave-private LDS (in-order per wave)
    #pragma unroll
    for (int nt = 0; nt < 4; ++nt)
      #pragma unroll
      for (int r = 0; r < 8; ++r)
        Ps[wave][(r + kb8) * 64 + nt * 16 + lr] = f2bf(s[nt][r]);

    #pragma unroll
    for (int kf = 0; kf < 2; ++kf) {
      int base = kf * 32 + kb8;
      FragBF pa;
      pa.q[0] = *(const uint4*)&Ps[wave][lr * 64 + base];
      pa.q[1] = *(const uint4*)&Ps[wave][lr * 64 + base + 16];
      #pragma unroll
      for (int j = 0; j < 4; ++j) {
        FragBF bv;
        int col = j * 16 + lr;  // hs index
        bv.q[0] = *(const uint4*)&Vt[col * 64 + base];
        bv.q[1] = *(const uint4*)&Vt[col * 64 + base + 16];
        o[j] = __builtin_amdgcn_wmma_f32_16x16x32_bf16(
            false, pa.v, false, bv.v, (short)0, o[j], false, false);
      }
    }
  }

  // epilogue: y[b, t, h*64 + hs] = O / l
  #pragma unroll
  for (int r = 0; r < 8; ++r) {
    float inv = 1.0f / lrun[r];
    int row = qb * 64 + wave * 16 + r + kb8;
    size_t base = ((size_t)b * cT + row) * cC + h * cHS;
    #pragma unroll
    for (int j = 0; j < 4; ++j)
      yb[base + j * 16 + lr] = f2bf(o[j][r] * inv);
  }
}

// ---------------------------------------------------------------------------
extern "C" void kernel_launch(void* const* d_in, const int* in_sizes, int n_in,
                              void* d_out, int out_size, void* d_ws, size_t ws_size,
                              hipStream_t stream)
{
  (void)in_sizes; (void)n_in; (void)out_size; (void)ws_size;
  const float* x      = (const float*)d_in[0];
  const float* memory = (const float*)d_in[1];
  const float* Wq     = (const float*)d_in[2];
  const float* bq     = (const float*)d_in[3];
  const float* Wkv    = (const float*)d_in[4];
  const float* bkv    = (const float*)d_in[5];
  const float* Wproj  = (const float*)d_in[6];
  const float* bproj  = (const float*)d_in[7];

  const size_t NT   = (size_t)cB * cT;   // 8192 tokens
  const size_t nAct = NT * cC;           // 6.29M elements

  char* base = (char*)d_ws;
  size_t off = 0;
  auto carve = [&](size_t elems) -> ushort_t* {
    ushort_t* r = (ushort_t*)(base + off);
    off += (elems * sizeof(ushort_t) + 255) & ~(size_t)255;
    return r;
  };
  ushort_t* xb   = carve(nAct);
  ushort_t* mb   = carve(nAct);
  ushort_t* wqb  = carve((size_t)cC * cC);
  ushort_t* wkvb = carve((size_t)cC * 2 * cC);
  ushort_t* wpb  = carve((size_t)cC * cC);
  ushort_t* qhb  = carve(nAct);   // [B,H,T,HS]
  ushort_t* khb  = carve(nAct);
  ushort_t* vhb  = carve(nAct);
  ushort_t* ybf  = carve(nAct);   // [B*T, C]

  // fp32 -> bf16
  cvt_f32_bf16<<<(int)(nAct / 1024), 256, 0, stream>>>(x,      xb,   (int)nAct);
  cvt_f32_bf16<<<(int)(nAct / 1024), 256, 0, stream>>>(memory, mb,   (int)nAct);
  cvt_f32_bf16<<<cC * cC / 1024,     256, 0, stream>>>(Wq,     wqb,  cC * cC);
  cvt_f32_bf16<<<cC * 2 * cC / 1024, 256, 0, stream>>>(Wkv,    wkvb, cC * 2 * cC);
  cvt_f32_bf16<<<cC * cC / 1024,     256, 0, stream>>>(Wproj,  wpb,  cC * cC);

  // Q = x Wq + bq  -> [B,H,T,HS] bf16
  gemm64<2><<<dim3(cC / 64, (int)(NT / 64)), 128, 0, stream>>>(
      xb, wqb, bq, qhb, nullptr, (int)NT, cC, cC);
  // KV = memory Wkv + bkv -> K,V [B,H,T,HS] bf16
  gemm64<3><<<dim3(2 * cC / 64, (int)(NT / 64)), 128, 0, stream>>>(
      mb, wkvb, bkv, khb, vhb, (int)NT, 2 * cC, cC);
  // attention -> y bf16 [B*T, C]
  attn64<<<dim3(cT / 64, cB * cH), 128, 0, stream>>>(qhb, khb, vhb, ybf);
  // out = y Wproj + bproj -> f32
  gemm64<0><<<dim3(cC / 64, (int)(NT / 64)), 128, 0, stream>>>(
      ybf, wpb, bproj, (float*)d_out, nullptr, (int)NT, cC, cC);
}
